// ResidualVectorQuantizer_38414187495828
// MI455X (gfx1250) — matrix operand order repaired and checked by reading, not compiled, not run on hardware
//
#include <hip/hip_runtime.h>

#define N_PTS 32768
#define DIM   256
#define NCODE 1024
#define NQ    4

typedef __attribute__((ext_vector_type(16))) __bf16 v16bf;
typedef __attribute__((ext_vector_type(8)))  float  v8f;
typedef __attribute__((ext_vector_type(4)))  int    v4i;

#if defined(__AMDGCN__) && __has_builtin(__builtin_amdgcn_global_load_async_to_lds_b128)
#define HAVE_ASYNC_LDS 1
#else
#define HAVE_ASYNC_LDS 0
#endif

#define AS1 __attribute__((address_space(1)))
#define AS3 __attribute__((address_space(3)))

__device__ __forceinline__ void wait_async0() {
#if __has_builtin(__builtin_amdgcn_s_wait_asynccnt)
  __builtin_amdgcn_s_wait_asynccnt(0);
#else
  asm volatile("s_wait_asynccnt 0x0" ::: "memory");
#endif
}

// ---------------- d_out layout (floats) ----------------
// x_q [N,256] | mean_losses [1] | all_indices [N,4] | all_distances [N,4,1024]
#define OFF_LOSS  ((size_t)N_PTS * DIM)
#define OFF_IDX   (OFF_LOSS + 1)
#define OFF_DIST  (OFF_IDX + (size_t)N_PTS * NQ)

// ---------------- workspace layout (bytes) ----------------
#define WS_Q    ((size_t)0)                                   // Sinkhorn Q [N,K] f32 (128 MB, L2-resident)
#define WS_RES  (WS_Q   + (size_t)N_PTS * NCODE * 4)          // residual f32 [N,256]
#define WS_RBF  (WS_RES + (size_t)N_PTS * DIM * 4)            // residual bf16 [N,256]
#define WS_CBF  (WS_RBF + (size_t)N_PTS * DIM * 2)            // codebooks bf16 [4,1024,256]
#define WS_RN   (WS_CBF + (size_t)NQ * NCODE * DIM * 2)       // row norms [N]
#define WS_CN   (WS_RN  + (size_t)N_PTS * 4)                  // code norms [4,1024]
#define WS_CSA  (WS_CN  + (size_t)NQ * NCODE * 4)             // column sums [1024]
#define WS_CSP  (WS_CSA + (size_t)NCODE * 4)                  // column-sum partials [256,1024]
#define WS_MM   (WS_CSP + (size_t)256 * NCODE * 4)            // ordered-uint max/min [2]
#define WS_LOSS (WS_MM  + 64)                                 // loss accumulator [1]

// ---------------- helpers ----------------
__device__ __forceinline__ unsigned short f2bf(float f) {
  unsigned u = __float_as_uint(f);
  unsigned r = ((u >> 16) & 1u) + 0x7FFFu;           // round-to-nearest-even
  return (unsigned short)((u + r) >> 16);
}
__device__ __forceinline__ unsigned f2o(float f) {   // monotonic float->uint
  unsigned u = __float_as_uint(f);
  return (u & 0x80000000u) ? ~u : (u | 0x80000000u);
}
__device__ __forceinline__ float o2f(unsigned u) {
  return (u & 0x80000000u) ? __uint_as_float(u & 0x7FFFFFFFu) : __uint_as_float(~u);
}

// ---------------- bf16 conversion + row sum-of-squares ----------------
__global__ __launch_bounds__(256) void rvq_prep_rows(const float* __restrict__ src,
                                                     float* __restrict__ res,
                                                     unsigned short* __restrict__ obf,
                                                     float* __restrict__ norms,
                                                     int copyRes) {
  __shared__ float red[256];
  const int n = blockIdx.x, d = threadIdx.x;
  const size_t i = (size_t)n * DIM + d;
  float v = src[i];
  if (copyRes) res[i] = v;
  obf[i] = f2bf(v);
  red[d] = v * v;
  __syncthreads();
  for (int s = 128; s > 0; s >>= 1) { if (d < s) red[d] += red[d + s]; __syncthreads(); }
  if (d == 0) norms[n] = red[0];
}

// ---------------- WMMA bf16 distance GEMM: d = rn + cn - 2 * r . c^T ----------------
#define BM 64
#define BN 64
#define BK 32
__global__ __launch_bounds__(256) void rvq_gemm_dist(const unsigned short* __restrict__ rbf,
                                                     const unsigned short* __restrict__ cbf,
                                                     const float* __restrict__ rn,
                                                     const float* __restrict__ cn,
                                                     float* __restrict__ gdist, int q) {
  __shared__ unsigned short ldsA[BM * BK];
  __shared__ unsigned short ldsB[BN * BK];
  __shared__ float rnT[BM];
  __shared__ float cnT[BN];

  const int t = threadIdx.x;
  const int lane = t & 31;
  const int wave = t >> 5;
  const int rowBase = blockIdx.y * BM;
  const int colBase = blockIdx.x * BN;

  if (t < BM)            rnT[t]      = rn[rowBase + t];
  else if (t < BM + BN)  cnT[t - BM] = cn[q * NCODE + colBase + (t - BM)];

  const int tr  = wave >> 1;        // tile-row (0..3), 16 rows each
  const int tc0 = (wave & 1) * 2;   // first tile-col (0 or 2), two 16-col tiles per wave

  v8f acc0 = {0.f, 0.f, 0.f, 0.f, 0.f, 0.f, 0.f, 0.f};
  v8f acc1 = acc0;

  const int lrow = t >> 2;          // 0..63
  const int lcol = (t & 3) * 8;     // 0,8,16,24 (8 bf16 = 16B per thread)
  const unsigned short* gA = rbf + (size_t)(rowBase + lrow) * DIM + lcol;
  const unsigned short* gB = cbf + ((size_t)q * NCODE + colBase + lrow) * DIM + lcol;
  unsigned short* lA = ldsA + lrow * BK + lcol;
  unsigned short* lB = ldsB + lrow * BK + lcol;

  for (int kk = 0; kk < DIM; kk += BK) {
    __syncthreads();
#if HAVE_ASYNC_LDS
    // gfx1250 direct global->LDS async copy (ASYNCcnt), no VGPR staging
    __builtin_amdgcn_global_load_async_to_lds_b128((AS1 v4i*)(gA + kk), (AS3 v4i*)lA, 0, 0);
    __builtin_amdgcn_global_load_async_to_lds_b128((AS1 v4i*)(gB + kk), (AS3 v4i*)lB, 0, 0);
    if (kk + BK < DIM) {
      __builtin_prefetch(gA + kk + BK, 0, 1);
      __builtin_prefetch(gB + kk + BK, 0, 1);
    }
    wait_async0();
#else
    *(uint4*)lA = *(const uint4*)(gA + kk);
    *(uint4*)lB = *(const uint4*)(gB + kk);
    if (kk + BK < DIM) {
      __builtin_prefetch(gA + kk + BK, 0, 1);
      __builtin_prefetch(gB + kk + BK, 0, 1);
    }
#endif
    __syncthreads();

    union { unsigned u[8]; v16bf v; } fa, fb0, fb1;
    {
      // A 16x32 bf16: lanes 0-15 K=0..15, lanes 16-31 K=16..31; pairs contiguous in K
      const int m  = tr * 16 + (lane & 15);
      const int kh = (lane >> 4) * 16;
      const unsigned short* rp = ldsA + m * BK;
      #pragma unroll
      for (int v = 0; v < 8; ++v) {
        const int k = 2 * v + kh;
        fa.u[v] = *(const unsigned*)(rp + k);
      }
    }
    {
      // B 32x16 bf16: lanes 0-15 K=0..15 (pairs per VGPR), lanes 16-31 K=16..31
      const int n0 = tc0 * 16 + (lane & 15);
      const int kb = (lane >> 4) * 16;
      const unsigned short* rp0 = ldsB + n0 * BK + kb;
      const unsigned short* rp1 = rp0 + 16 * BK;
      #pragma unroll
      for (int v = 0; v < 8; ++v) {
        fb0.u[v] = *(const unsigned*)(rp0 + 2 * v);
        fb1.u[v] = *(const unsigned*)(rp1 + 2 * v);
      }
    }
    acc0 = __builtin_amdgcn_wmma_f32_16x16x32_bf16(false, fa.v, false, fb0.v, (short)0, acc0, false, false);
    acc1 = __builtin_amdgcn_wmma_f32_16x16x32_bf16(false, fa.v, false, fb1.v, (short)0, acc1, false, false);
  }

  // C/D layout: VGPR r -> M = tr*16 + r + 8*(lane>>4); N = lane&15
  const int colL0 = tc0 * 16 + (lane & 15);
  const int colL1 = colL0 + 16;
  const int hs = (lane >> 4) * 8;
  const float c0 = cnT[colL0];
  const float c1 = cnT[colL1];
  #pragma unroll
  for (int r = 0; r < 8; ++r) {
    const int rowL = tr * 16 + r + hs;
    const float rv = rnT[rowL];
    float* dst = gdist + (size_t)(rowBase + rowL) * (NQ * NCODE) + (size_t)q * NCODE + colBase;
    dst[colL0] = rv + c0 - 2.0f * acc0[r];
    dst[colL1] = rv + c1 - 2.0f * acc1[r];
  }
}

// ---------------- global max/min over one stage's distance slice ----------------
__global__ __launch_bounds__(256) void rvq_mm_init(unsigned* mm) {
  if (threadIdx.x == 0) { mm[0] = 0u; mm[1] = 0xFFFFFFFFu; }
}
__global__ __launch_bounds__(256) void rvq_minmax(const float* __restrict__ gdist, unsigned* mm, int q) {
  const size_t total = (size_t)N_PTS * NCODE;
  float vmax = -3.4e38f, vmin = 3.4e38f;
  for (size_t i = (size_t)blockIdx.x * blockDim.x + threadIdx.x; i < total;
       i += (size_t)gridDim.x * blockDim.x) {
    const size_t n = i >> 10;
    const int k = (int)(i & 1023);
    const float v = gdist[n * (NQ * NCODE) + (size_t)q * NCODE + k];
    vmax = fmaxf(vmax, v);
    vmin = fminf(vmin, v);
  }
  for (int off = 16; off; off >>= 1) {
    vmax = fmaxf(vmax, __shfl_xor(vmax, off, 32));
    vmin = fminf(vmin, __shfl_xor(vmin, off, 32));
  }
  if ((threadIdx.x & 31) == 0) {
    atomicMax(mm + 0, f2o(vmax));
    atomicMin(mm + 1, f2o(vmin));
  }
}

// ---------------- Q = exp(-(dc - dc_min)/eps), fused initial column sums ----------------
// Launch with grid 2048x256 so the stride (524288) is a multiple of 1024:
// each thread touches a single fixed column k.
__global__ __launch_bounds__(256) void rvq_exp(const float* __restrict__ gdist, float* __restrict__ Q,
                                               float* __restrict__ csPart,
                                               const unsigned* __restrict__ mm, float invEps, int q) {
  const float mx = o2f(mm[0]), mn = o2f(mm[1]);
  const float mid = 0.5f * (mx + mn);
  const float amp = mx - mid + 1e-5f;
  const float dcmin = (mn - mid) / amp;
  const size_t total = (size_t)N_PTS * NCODE;
  const size_t stride = (size_t)gridDim.x * blockDim.x;
  const size_t i0 = (size_t)blockIdx.x * blockDim.x + threadIdx.x;
  const int k = (int)(i0 & 1023);
  float csum = 0.f;
  for (size_t i = i0; i < total; i += stride) {
    const size_t n = i >> 10;
    const float v = gdist[n * (NQ * NCODE) + (size_t)q * NCODE + k];
    const float dc = (v - mid) / amp;
    const float e = __expf(-(dc - dcmin) * invEps);
    Q[i] = e;
    csum += e;
  }
  atomicAdd(csPart + (size_t)(blockIdx.x & 255) * NCODE + k, csum);
}

// ---------------- fold partial column sums -> csOut, re-zero partials ----------------
__global__ __launch_bounds__(256) void rvq_colreduce(float* __restrict__ part, float* __restrict__ csOut) {
  const int c = blockIdx.x * 256 + threadIdx.x;   // grid 4 -> 1024 columns
  float s = 0.f;
  for (int b = 0; b < 256; ++b) {
    s += part[(size_t)b * NCODE + c];
    part[(size_t)b * NCODE + c] = 0.f;
  }
  csOut[c] = s;
}

// ---------------- fused Sinkhorn step: col-scale, row-sum, row-scale, next col-sums, opt argmax ----------------
__global__ __launch_bounds__(256) void rvq_rownorm(float* __restrict__ Q,
                                                   const float* __restrict__ csIn,
                                                   float* __restrict__ csPart,
                                                   float* __restrict__ idxOut, int q, int writeIdx) {
  __shared__ float cs[NCODE];
  __shared__ float acc[NCODE];
  const int t = threadIdx.x;
  for (int i = t; i < NCODE; i += 256) { cs[i] = csIn[i]; acc[i] = 0.f; }
  __syncthreads();

  const int row = blockIdx.x * 8 + (t >> 5);
  const int lane = t & 31;
  float* p = Q + (size_t)row * NCODE;
  float vals[32];
  float rs = 0.f;
  #pragma unroll
  for (int j = 0; j < 32; ++j) {
    const int c = lane + j * 32;
    const float v = p[c] / (cs[c] * (float)NCODE);
    vals[j] = v;
    rs += v;
  }
  for (int off = 16; off; off >>= 1) rs += __shfl_xor(rs, off, 32);
  const float inv = 1.0f / (rs * (float)N_PTS);
  #pragma unroll
  for (int j = 0; j < 32; ++j) {
    const int c = lane + j * 32;
    const float o = vals[j] * inv;
    p[c] = o;
    atomicAdd(&acc[c], o);           // ds_add_f32, next iteration's column sums
  }
  if (writeIdx) {                    // argmax: row scale is positive, vals order == final order
    float best = -3.4e38f; int bi = 0;
    #pragma unroll
    for (int j = 0; j < 32; ++j) {
      if (vals[j] > best) { best = vals[j]; bi = lane + j * 32; }
    }
    for (int off = 16; off; off >>= 1) {
      const float ov = __shfl_xor(best, off, 32);
      const int oi = __shfl_xor(bi, off, 32);
      if (ov > best || (ov == best && oi < bi)) { best = ov; bi = oi; }
    }
    if (lane == 0) idxOut[(size_t)row * NQ + q] = (float)bi;
  }
  __syncthreads();
  float* prow = csPart + (size_t)(blockIdx.x & 255) * NCODE;
  for (int i = t; i < NCODE; i += 256) atomicAdd(&prow[i], acc[i]);
}

// ---------------- per-row argmin for stage 0 (first occurrence on ties) ----------------
__global__ __launch_bounds__(256) void rvq_argsel(const float* __restrict__ src, long rowStride,
                                                  int colOffset, float* __restrict__ idxOut, int q) {
  const int row = blockIdx.x * 8 + (threadIdx.x >> 5);
  const int lane = threadIdx.x & 31;
  const float* p = src + (size_t)row * rowStride + colOffset;
  float best = 3.4e38f;
  int bi = 0;
  for (int c = lane; c < NCODE; c += 32) {
    const float v = p[c];
    if (v < best) { best = v; bi = c; }
  }
  for (int off = 16; off; off >>= 1) {
    const float ov = __shfl_xor(best, off, 32);
    const int oi = __shfl_xor(bi, off, 32);
    if (ov < best || (ov == best && oi < bi)) { best = ov; bi = oi; }
  }
  if (lane == 0) idxOut[(size_t)row * NQ + q] = (float)bi;
}

// ---------------- gather + residual update + loss SSE ----------------
__global__ __launch_bounds__(256) void rvq_update(const float* __restrict__ cbq,
                                                  const float* __restrict__ idxF,
                                                  float* __restrict__ res,
                                                  float* lossAcc, int q) {
  __shared__ float red[256];
  const int n = blockIdx.x, d = threadIdx.x;
  const int idx = (int)idxF[(size_t)n * NQ + q];
  const float c = cbq[(size_t)idx * DIM + d];
  const size_t i = (size_t)n * DIM + d;
  const float r = res[i];
  const float diff = c - r;          // (x_q - residual)
  res[i] = r - c;                    // next residual
  red[d] = diff * diff;
  __syncthreads();
  for (int s = 128; s > 0; s >>= 1) { if (d < s) red[d] += red[d + s]; __syncthreads(); }
  if (d == 0) atomicAdd(lossAcc, red[0]);
}

// ---------------- finalize ----------------
__global__ __launch_bounds__(256) void rvq_xq(const float* __restrict__ x,
                                              const float* __restrict__ res,
                                              float* __restrict__ out) {
  const size_t i = (size_t)blockIdx.x * blockDim.x + threadIdx.x;
  out[i] = x[i] - res[i];            // x_q = x - final residual (straight-through forward)
}
__global__ void rvq_loss(const float* lossAcc, float* out) {
  out[0] = 1.25f * lossAcc[0] / (float)((size_t)NQ * N_PTS * DIM);
}
__global__ __launch_bounds__(256) void rvq_zero(float* p, int n) {
  const int i = blockIdx.x * blockDim.x + threadIdx.x;
  if (i < n) p[i] = 0.f;
}

extern "C" void kernel_launch(void* const* d_in, const int* in_sizes, int n_in,
                              void* d_out, int out_size, void* d_ws, size_t ws_size,
                              hipStream_t stream) {
  const float* x  = (const float*)d_in[0];
  const float* cb = (const float*)d_in[1];
  float* out = (float*)d_out;
  char* ws = (char*)d_ws;

  float*          Q       = (float*)(ws + WS_Q);
  float*          res     = (float*)(ws + WS_RES);
  unsigned short* rbf     = (unsigned short*)(ws + WS_RBF);
  unsigned short* cbf     = (unsigned short*)(ws + WS_CBF);
  float*          rn      = (float*)(ws + WS_RN);
  float*          cn      = (float*)(ws + WS_CN);
  float*          csA     = (float*)(ws + WS_CSA);
  float*          csPart  = (float*)(ws + WS_CSP);
  unsigned*       mm      = (unsigned*)(ws + WS_MM);
  float*          lossAcc = (float*)(ws + WS_LOSS);

  float* gdist = out + OFF_DIST;
  float* gidx  = out + OFF_IDX;

  const float invEps[NQ] = {0.f, 1.0f / 0.003f, 1.0f / 0.01f, 1.0f / 0.1f};

  // codebooks -> bf16 + ||c||^2 ; zero accumulators/partials
  rvq_prep_rows<<<NQ * NCODE, 256, 0, stream>>>(cb, nullptr, cbf, cn, 0);
  rvq_zero<<<1, 256, 0, stream>>>(lossAcc, 1);
  rvq_zero<<<1024, 256, 0, stream>>>(csPart, 256 * NCODE);

  for (int q = 0; q < NQ; ++q) {
    rvq_prep_rows<<<N_PTS, 256, 0, stream>>>(q == 0 ? x : res, res, rbf, rn, q == 0 ? 1 : 0);
    rvq_gemm_dist<<<dim3(NCODE / BN, N_PTS / BM), 256, 0, stream>>>(rbf, cbf, rn, cn, gdist, q);

    if (q > 0) {
      rvq_mm_init<<<1, 32, 0, stream>>>(mm);
      rvq_minmax<<<2048, 256, 0, stream>>>(gdist, mm, q);
      rvq_exp<<<2048, 256, 0, stream>>>(gdist, Q, csPart, mm, invEps[q], q);
      rvq_colreduce<<<4, 256, 0, stream>>>(csPart, csA);
      for (int it = 0; it < 20; ++it) {
        rvq_rownorm<<<N_PTS / 8, 256, 0, stream>>>(Q, csA, csPart, gidx, q, it == 19);
        rvq_colreduce<<<4, 256, 0, stream>>>(csPart, csA);  // also re-zeros partials
      }
    } else {
      rvq_argsel<<<N_PTS / 8, 256, 0, stream>>>(gdist, NQ * NCODE, q * NCODE, gidx, q);
    }

    rvq_update<<<N_PTS, 256, 0, stream>>>(cb + (size_t)q * NCODE * DIM, gidx, res, lossAcc, q);
  }

  rvq_xq<<<(N_PTS * DIM) / 256, 256, 0, stream>>>(x, res, out);
  rvq_loss<<<1, 1, 0, stream>>>(lossAcc, out + OFF_LOSS);
}
